// MultiQueryAttention_53128745452310
// MI455X (gfx1250) — matrix operand order repaired and checked by reading, not compiled, not run on hardware
//
#include <hip/hip_runtime.h>

// ---------------------------------------------------------------------------
// MQA forward for MI455X (gfx1250, wave32, WMMA bf16 16x16x32, f32 accum)
// B=2, S=2048, HID=2048, NH=16, HD=128
// Async-to-LDS double-buffered GEMM + TDM-staged flash attention.
// ---------------------------------------------------------------------------

typedef __bf16 bf16;
typedef __attribute__((ext_vector_type(16))) __bf16 v16bf;
typedef __attribute__((ext_vector_type(8)))  float  v8f;
typedef __attribute__((ext_vector_type(4)))  unsigned int u32x4;
typedef __attribute__((ext_vector_type(8)))  int          i32x8;
typedef __attribute__((ext_vector_type(4)))  int          i32x4;

union FragU { v16bf v; uint4 u[2]; };
union Pack8 { uint4 u; bf16 e[8]; };

#if defined(__HIP_DEVICE_COMPILE__) && __has_builtin(__builtin_amdgcn_tensor_load_to_lds)
#define HAS_TDM 1
#else
#define HAS_TDM 0
#endif

__device__ inline v8f vzero() {
    v8f z;
    #pragma unroll
    for (int i = 0; i < 8; ++i) z[i] = 0.0f;
    return z;
}

__device__ inline v8f wmma_bf16(v16bf a, v16bf b, v8f c) {
    return __builtin_amdgcn_wmma_f32_16x16x32_bf16(
        false, a, false, b, (short)0, c, false, false);
}

// 16-byte async copy global -> LDS (per-lane addresses, ASYNCcnt tracked).
__device__ inline void async_copy16(void* lds_dst, const void* gsrc) {
    unsigned lds = (unsigned)(unsigned long long)(uintptr_t)lds_dst; // low 32 = LDS offset
    unsigned long long g = (unsigned long long)(uintptr_t)gsrc;
    asm volatile("global_load_async_to_lds_b128 %0, %1, off"
                 :: "v"(lds), "v"(g) : "memory");
}

__device__ inline void wait_async0() {
    asm volatile("s_wait_asynccnt 0x0" ::: "memory");
}

// A fragment: 16x32 bf16 tile in LDS, row stride `ld` elems.
__device__ inline v16bf load_frag_a(const bf16* p, int ld) {
    int lane = threadIdx.x & 31;
    int row  = lane & 15;
    int kb   = (lane >> 4) * 8;
    FragU f;
    f.u[0] = *(const uint4*)(p + row * ld + kb);
    f.u[1] = *(const uint4*)(p + row * ld + 16 + kb);
    return f.v;
}

// B fragment: 32x16 tile stored as N rows x 32 contiguous K (row stride ld).
__device__ inline v16bf load_frag_b(const bf16* p, int ld) {
    int lane = threadIdx.x & 31;
    int col  = lane & 15;
    int kb   = (lane >> 4) * 8;
    FragU f;
    f.u[0] = *(const uint4*)(p + col * ld + kb);
    f.u[1] = *(const uint4*)(p + col * ld + 16 + kb);
    return f.v;
}

// ---------------------------------------------------------------------------
// f32 -> bf16 conversion
// ---------------------------------------------------------------------------
__global__ void cvt_f32_bf16(const float* __restrict__ in,
                             bf16* __restrict__ out, int n) {
    int i = blockIdx.x * 256 + threadIdx.x;
    if (i < n) out[i] = (bf16)in[i];
}

// ---------------------------------------------------------------------------
// C[M,N] = A[M,K](bf16) * Bt[N,K](bf16)^T + bias[N]
// 128x128 tile / 256 threads; 8 waves x (64x32); K-step 32.
// Double-buffered LDS, async-to-LDS prefetch, one barrier per K-step.
// ---------------------------------------------------------------------------
#define LDT 40  // padded K stride in elems (80 B, 16B-aligned rows)

__global__ __launch_bounds__(256) void gemm_bf16(
    const bf16* __restrict__ A, const bf16* __restrict__ Bt,
    const float* __restrict__ bias,
    float* __restrict__ Cf, bf16* __restrict__ Cb,
    int M, int N, int K) {
    __shared__ __align__(16) bf16 As[2][128 * LDT];
    __shared__ __align__(16) bf16 Bs[2][128 * LDT];

    int tid  = threadIdx.x;
    int wave = tid >> 5;
    int lane = tid & 31;
    int m0 = blockIdx.y * 128;
    int n0 = blockIdx.x * 128;
    int wm = (wave >> 2) * 64;
    int wn = (wave & 3) * 32;

    v8f acc[4][2];
    #pragma unroll
    for (int i = 0; i < 4; ++i)
        #pragma unroll
        for (int j = 0; j < 2; ++j) acc[i][j] = vzero();

    int rowA = tid >> 1;
    int half = (tid & 1) * 16;
    const bf16* aPtr = A  + (size_t)(m0 + rowA) * K + half;
    const bf16* bPtr = Bt + (size_t)(n0 + rowA) * K + half;
    unsigned ldsOff = rowA * LDT + half;

    // prime buffer 0
    async_copy16(&As[0][ldsOff], aPtr);
    async_copy16(&Bs[0][ldsOff], bPtr);
    wait_async0();
    __syncthreads();

    int cur = 0;
    for (int k0 = 0; k0 < K; k0 += 32) {
        int nxt = cur ^ 1;
        if (k0 + 32 < K) {
            // prefetch next tile while computing on current
            async_copy16(&As[nxt][ldsOff], aPtr + k0 + 32);
            async_copy16(&Bs[nxt][ldsOff], bPtr + k0 + 32);
        }

        v16bf bf0 = load_frag_b(&Bs[cur][(wn + 0) * LDT], LDT);
        v16bf bf1 = load_frag_b(&Bs[cur][(wn + 16) * LDT], LDT);
        #pragma unroll
        for (int ms = 0; ms < 4; ++ms) {
            v16bf af = load_frag_a(&As[cur][(wm + ms * 16) * LDT], LDT);
            acc[ms][0] = wmma_bf16(af, bf0, acc[ms][0]);
            acc[ms][1] = wmma_bf16(af, bf1, acc[ms][1]);
        }

        wait_async0();      // own prefetch done (overlapped with WMMAs)
        __syncthreads();    // everyone's prefetch visible; cur free for reuse
        cur = nxt;
    }

    int col = lane & 15;
    int hi  = lane >> 4;
    #pragma unroll
    for (int ms = 0; ms < 4; ++ms) {
        #pragma unroll
        for (int ns = 0; ns < 2; ++ns) {
            int n = n0 + wn + ns * 16 + col;
            float bv = bias ? bias[n] : 0.0f;
            #pragma unroll
            for (int r = 0; r < 8; ++r) {
                int m = m0 + wm + ms * 16 + r + hi * 8;
                float val = acc[ms][ns][r] + bv;
                if (Cf) Cf[(size_t)m * N + n] = val;
                else    Cb[(size_t)m * N + n] = (bf16)val;
            }
        }
    }
}

// ---------------------------------------------------------------------------
// Flash attention, single KV head shared by 16 query heads.
// Q: (B,S,2048) bf16; Kg,Vg: (B,S,128) bf16; mask: (B,1,1,S) int32
// Out: (B,S,2048) bf16. Block = (b,h, 128 q rows); 8 waves x 16 rows.
// Q tile staged by TDM (tensor_load_to_lds), K tiles by async-to-LDS.
// ---------------------------------------------------------------------------
#define SEQ 2048
#define HDIM 128

__global__ __launch_bounds__(256) void mqa_attention(
    const bf16* __restrict__ Q, const bf16* __restrict__ Kg,
    const bf16* __restrict__ Vg, const int* __restrict__ mask,
    bf16* __restrict__ Out) {
    __shared__ __align__(16) bf16 Qs[128 * 136];   // 34816 B (row pad: 256B+16B)
    __shared__ __align__(16) bf16 Ks[32 * 136];    //  8704 B
    __shared__ __align__(16) bf16 Vt[128 * 40];    // 10240 B (V^T: [d][key])
    __shared__ __align__(16) bf16 Ps[8 * 16 * 32]; //  8192 B (per-wave P scratch)
    __shared__ int mk[32];

    int tid  = threadIdx.x;
    int wave = tid >> 5;
    int lane = tid & 31;
    int bh = blockIdx.x;
    int bb = bh >> 4;
    int hh = bh & 15;
    int q0 = blockIdx.y * 128;

    const size_t qgbase = (size_t)bb * SEQ * 2048 + (size_t)q0 * 2048 + hh * HDIM;

#if HAS_TDM
    // Stage the 128x128 bf16 Q tile with the Tensor Data Mover.
    if (tid < 32) {  // wave 0 issues the DMA
        unsigned long long ga = (unsigned long long)(uintptr_t)(Q + qgbase);
        unsigned lds = (unsigned)(uintptr_t)(void*)Qs;
        u32x4 g0;
        g0[0] = 1u;                                  // count=1, user descriptor
        g0[1] = lds;                                 // lds_addr
        g0[2] = (unsigned)(ga & 0xffffffffu);        // global_addr[31:0]
        g0[3] = (unsigned)((ga >> 32) & 0x01ffffffu) // global_addr[56:32]
                | 0x80000000u;                       // type=2 ("image")
        unsigned td0 = 2048u;            // tensor_dim0 (elems)
        unsigned td1 = 1u << 20;         // tensor_dim1 (large, no OOB)
        unsigned tile0 = 128u, tile1 = 128u;
        unsigned long long st0 = 2048ull; // tensor_dim0_stride (elems)
        i32x8 g1;
        g1[0] = (int)((1u << 16)         // data_size = 2 bytes
                | (1u << 20)             // pad_enable
                | (5u << 22)             // pad_interval: 64 DWORDs (256 B)
                | (3u << 25));           // pad_amount: 4 DWORDs (16 B)
        g1[1] = (int)((td0 & 0xffffu) << 16);
        g1[2] = (int)(((td0 >> 16) & 0xffffu) | ((td1 & 0xffffu) << 16));
        g1[3] = (int)(((td1 >> 16) & 0xffffu) | ((tile0 & 0xffffu) << 16));
        g1[4] = (int)(tile1 & 0xffffu);  // tile_dim1; tile_dim2=0
        g1[5] = (int)(unsigned)(st0 & 0xffffffffull);
        g1[6] = (int)(unsigned)((st0 >> 32) & 0xffffull);
        g1[7] = 0;
        i32x4 gz4 = {0, 0, 0, 0};
        i32x8 gz8 = {0, 0, 0, 0, 0, 0, 0, 0};
        __builtin_amdgcn_tensor_load_to_lds(g0, g1, gz4, gz4, gz8, 0);
        __builtin_amdgcn_s_wait_tensorcnt((short)0);
    }
#else
    for (int i = tid; i < 128 * 16; i += 256) {
        int row = i >> 4;
        int c   = (i & 15) * 8;
        *(uint4*)(Qs + row * 136 + c) =
            *(const uint4*)(Q + qgbase + (size_t)row * 2048 + c);
    }
#endif
    __syncthreads();

    float mrow[8], lrow[8];
    v8f acc[8];
    #pragma unroll
    for (int r = 0; r < 8; ++r) { mrow[r] = -1e30f; lrow[r] = 0.0f; acc[r] = vzero(); }

    const bf16* kptr = Kg + (size_t)bb * SEQ * HDIM;
    const bf16* vptr = Vg + (size_t)bb * SEQ * HDIM;
    const int*  mptr = mask + (size_t)bb * SEQ;
    const float scale = 0.08838834764831845f;  // 1/sqrt(128)

    int col = lane & 15;
    int hi  = lane >> 4;
    bf16* prow = Ps + wave * 16 * 32;

    // per-thread K-staging coords: 2 chunks of 16B (32 rows x 16 chunks)
    int kc0 = tid * 2, kc1 = tid * 2 + 1;
    int kr0 = kc0 >> 4, ko0 = (kc0 & 15) * 8;
    int kr1 = kc1 >> 4, ko1 = (kc1 & 15) * 8;

    for (int kb = 0; kb < SEQ; kb += 32) {
        // K block via async-to-LDS (no VGPR round trip)
        async_copy16(Ks + kr0 * 136 + ko0, kptr + (size_t)(kb + kr0) * HDIM + ko0);
        async_copy16(Ks + kr1 * 136 + ko1, kptr + (size_t)(kb + kr1) * HDIM + ko1);
        // V block transposed through VGPRs: Vt[d][key]
        for (int i = tid; i < 32 * 16; i += 256) {
            int key = i >> 4;
            int c   = (i & 15) * 8;
            Pack8 t;
            t.u = *(const uint4*)(vptr + (size_t)(kb + key) * HDIM + c);
            #pragma unroll
            for (int j = 0; j < 8; ++j) Vt[(c + j) * 40 + key] = t.e[j];
        }
        if (tid < 32) mk[tid] = mptr[kb + tid];
        wait_async0();
        __syncthreads();

        // scores: 16 q-rows x 32 keys (two 16x16 tiles)
        int qrow = wave * 16;
        v8f s0 = vzero(), s1 = vzero();
        #pragma unroll
        for (int dc = 0; dc < 4; ++dc) {
            v16bf af  = load_frag_a(Qs + qrow * 136 + dc * 32, 136);
            v16bf kf0 = load_frag_b(Ks + 0 * 136 + dc * 32, 136);
            v16bf kf1 = load_frag_b(Ks + 16 * 136 + dc * 32, 136);
            s0 = wmma_bf16(af, kf0, s0);
            s1 = wmma_bf16(af, kf1, s1);
        }

        bool ok0 = mk[col] != 0;
        bool ok1 = mk[16 + col] != 0;

        // online softmax; row r+8*hi, its 16 cols live in one 16-lane group
        #pragma unroll
        for (int r = 0; r < 8; ++r) {
            float v0 = ok0 ? s0[r] * scale : -1e30f;
            float v1 = ok1 ? s1[r] * scale : -1e30f;
            float mx = fmaxf(v0, v1);
            mx = fmaxf(mx, __shfl_xor(mx, 1, 16));
            mx = fmaxf(mx, __shfl_xor(mx, 2, 16));
            mx = fmaxf(mx, __shfl_xor(mx, 4, 16));
            mx = fmaxf(mx, __shfl_xor(mx, 8, 16));
            float mn = fmaxf(mrow[r], mx);
            float p0 = __expf(v0 - mn);
            float p1 = __expf(v1 - mn);
            float corr = (mrow[r] > -9e29f) ? __expf(mrow[r] - mn) : 0.0f;
            float rs = p0 + p1;
            rs += __shfl_xor(rs, 1, 16);
            rs += __shfl_xor(rs, 2, 16);
            rs += __shfl_xor(rs, 4, 16);
            rs += __shfl_xor(rs, 8, 16);
            lrow[r] = lrow[r] * corr + rs;
            mrow[r] = mn;
            #pragma unroll
            for (int d = 0; d < 8; ++d) acc[d][r] *= corr;
            int rr = r + hi * 8;
            prow[rr * 32 + col]      = (bf16)p0;
            prow[rr * 32 + 16 + col] = (bf16)p1;
        }

        // out += P(16x32) @ V(32x128)
        v16bf pa = load_frag_a(prow, 32);
        #pragma unroll
        for (int d = 0; d < 8; ++d) {
            v16bf vb = load_frag_b(Vt + (d * 16) * 40, 40);
            acc[d] = wmma_bf16(pa, vb, acc[d]);
        }
        __syncthreads();
    }

    #pragma unroll
    for (int r = 0; r < 8; ++r) {
        float inv = lrow[r] > 0.0f ? 1.0f / lrow[r] : 0.0f;
        int rr = r + hi * 8;
        size_t orow = (size_t)bb * SEQ * 2048 +
                      (size_t)(q0 + wave * 16 + rr) * 2048 + hh * HDIM;
        #pragma unroll
        for (int d = 0; d < 8; ++d)
            Out[orow + d * 16 + col] = (bf16)(acc[d][r] * inv);
    }
}

// ---------------------------------------------------------------------------
// launch
// ---------------------------------------------------------------------------
extern "C" void kernel_launch(void* const* d_in, const int* in_sizes, int n_in,
                              void* d_out, int out_size, void* d_ws, size_t ws_size,
                              hipStream_t stream) {
    (void)in_sizes; (void)n_in; (void)out_size; (void)ws_size;
    const int B = 2, S = 2048, HID = 2048, HD = 128;
    const int M = B * S;

    const float* X  = (const float*)d_in[0];
    const int*   Am = (const int*)  d_in[1];
    const float* Wq = (const float*)d_in[2];
    const float* bq = (const float*)d_in[3];
    const float* Wk = (const float*)d_in[4];
    const float* bk = (const float*)d_in[5];
    const float* Wv = (const float*)d_in[6];
    const float* bv = (const float*)d_in[7];
    const float* Wo = (const float*)d_in[8];
    const float* bo = (const float*)d_in[9];
    float* out = (float*)d_out;

    char* ws = (char*)d_ws;
    bf16* Xbf  = (bf16*)ws; ws += (size_t)M * HID * 2;
    bf16* Qbf  = (bf16*)ws; ws += (size_t)M * HID * 2;
    bf16* Abf  = (bf16*)ws; ws += (size_t)M * HID * 2;
    bf16* Wqbf = (bf16*)ws; ws += (size_t)HID * HID * 2;
    bf16* Wobf = (bf16*)ws; ws += (size_t)HID * HID * 2;
    bf16* Wkbf = (bf16*)ws; ws += (size_t)HD * HID * 2;
    bf16* Wvbf = (bf16*)ws; ws += (size_t)HD * HID * 2;
    bf16* Kbf  = (bf16*)ws; ws += (size_t)M * HD * 2;
    bf16* Vbf  = (bf16*)ws; ws += (size_t)M * HD * 2;

    auto cvt = [&](const float* src, bf16* dst, int n) {
        cvt_f32_bf16<<<(n + 255) / 256, 256, 0, stream>>>(src, dst, n);
    };
    cvt(X,  Xbf,  M * HID);
    cvt(Wq, Wqbf, HID * HID);
    cvt(Wk, Wkbf, HD * HID);
    cvt(Wv, Wvbf, HD * HID);
    cvt(Wo, Wobf, HID * HID);

    dim3 blk(256);
    gemm_bf16<<<dim3(HID / 128, M / 128), blk, 0, stream>>>(
        Xbf, Wqbf, bq, nullptr, Qbf, M, HID, HID);
    gemm_bf16<<<dim3(1, M / 128), blk, 0, stream>>>(
        Xbf, Wkbf, bk, nullptr, Kbf, M, HD, HID);
    gemm_bf16<<<dim3(1, M / 128), blk, 0, stream>>>(
        Xbf, Wvbf, bv, nullptr, Vbf, M, HD, HID);
    mqa_attention<<<dim3(B * 16, S / 128), blk, 0, stream>>>(
        Qbf, Kbf, Vbf, Am, Abf);
    gemm_bf16<<<dim3(HID / 128, M / 128), blk, 0, stream>>>(
        Abf, Wobf, bo, out, nullptr, M, HID, HID);
}
